// TarCommAgent_45689862095602
// MI455X (gfx1250) — compile-verified, matrix operand order checked by reading, not run on hardware
//
#include <hip/hip_runtime.h>
#include <hip/hip_bf16.h>
#include <math.h>

// Problem constants (match reference)
#define N_AG  2048
#define OBS   512
#define HID   2048
#define NACT  32
#define PASSES 4
#define QKD   16

// GEMM tiling
#define BM 128
#define BN 128
#define BK 64
#define LDSS 72   // padded LDS row stride in ushorts (64 + 8) -> 144B, 16B aligned

// ---- optional CDNA5 async global->LDS path (guarded; sync fallback) ----
#if defined(__has_builtin)
#  if __has_builtin(__builtin_amdgcn_global_load_async_to_lds_b128) && \
      __has_builtin(__builtin_amdgcn_s_wait_asynccnt)
#    define USE_ASYNC_LDS 1
#  endif
#endif
#ifndef USE_ASYNC_LDS
#  define USE_ASYNC_LDS 0
#endif

// exact vector spelling the builtin expects: 'int __attribute__((vector_size(16)))'
typedef int v4i __attribute__((vector_size(16)));
typedef __attribute__((address_space(1))) v4i gv4i_t;   // global (printed '__device__')
typedef __attribute__((address_space(3))) v4i lv4i_t;   // LDS    (printed '__shared__')
#define GAS(p) ((gv4i_t*)(void*)(p))
#define LAS(p) ((lv4i_t*)(void*)(p))

typedef __attribute__((ext_vector_type(8)))  float   v8f;
typedef __attribute__((ext_vector_type(8)))  __bf16  v8bf;
typedef __attribute__((ext_vector_type(16))) __bf16  v16bf;

__device__ __forceinline__ unsigned short f2bf(float f) {
  union { float f; unsigned u; } v; v.f = f;
  unsigned r = v.u + 0x7FFFu + ((v.u >> 16) & 1u);   // round-to-nearest-even
  return (unsigned short)(r >> 16);
}
__device__ __forceinline__ float bf2f(unsigned short b) {
  union { unsigned u; float f; } v; v.u = ((unsigned)b) << 16;
  return v.f;
}

// float32 -> bf16 conversion (grid-stride)
__global__ void cvt_bf16_kernel(const float* __restrict__ in,
                                unsigned short* __restrict__ out, size_t n) {
  size_t i = (size_t)blockIdx.x * blockDim.x + threadIdx.x;
  size_t stride = (size_t)gridDim.x * blockDim.x;
  for (; i < n; i += stride) out[i] = f2bf(in[i]);
}

// Stage one BM x BK A-tile and BN x BK B-tile into LDS (Bs laid out [n][k]).
__device__ __forceinline__
void stage_ab(unsigned short* __restrict__ as,
              unsigned short* __restrict__ bs,
              const unsigned short* __restrict__ A,
              const unsigned short* __restrict__ B,
              int bT, int m0, int n0, int kt, int K, int Nm, int tid)
{
  // A tile: 128 rows x 64 k = 8192 ushorts; 16B per thread-iter, 4 iters.
  #pragma unroll
  for (int it = 0; it < 4; ++it) {
    int idx = it * 256 + tid;            // 0..1023
    int row = idx >> 3;                  // 0..127
    int kq  = (idx & 7) << 3;            // 0,8,...,56
    const unsigned short* g = &A[(size_t)(m0 + row) * K + kt + kq];
    unsigned short* l = &as[row * LDSS + kq];
#if USE_ASYNC_LDS
    __builtin_amdgcn_global_load_async_to_lds_b128(GAS(g), LAS(l), 0, 0);
#else
    *(uint4*)l = *(const uint4*)g;
#endif
  }
  if (!bT) {
    #pragma unroll
    for (int it = 0; it < 4; ++it) {
      int idx = it * 256 + tid;
      int row = idx >> 3;
      int kq  = (idx & 7) << 3;
      const unsigned short* g = &B[(size_t)(n0 + row) * K + kt + kq];
      unsigned short* l = &bs[row * LDSS + kq];
#if USE_ASYNC_LDS
      __builtin_amdgcn_global_load_async_to_lds_b128(GAS(g), LAS(l), 0, 0);
#else
      *(uint4*)l = *(const uint4*)g;
#endif
    }
  } else {
    // B is (K x Nm): coalesced 8B global reads along n, scalar LDS scatter.
    #pragma unroll
    for (int it = 0; it < 8; ++it) {
      int idx = it * 256 + tid;          // 0..2047
      int k   = idx >> 5;                // 0..63
      int ng  = (idx & 31) << 2;         // 0..124
      uint2 d = *(const uint2*)(&B[(size_t)(kt + k) * Nm + n0 + ng]);
      const unsigned short* ds = (const unsigned short*)&d;
      bs[(ng + 0) * LDSS + k] = ds[0];
      bs[(ng + 1) * LDSS + k] = ds[1];
      bs[(ng + 2) * LDSS + k] = ds[2];
      bs[(ng + 3) * LDSS + k] = ds[3];
    }
  }
}

// Generic bf16 WMMA GEMM:
//   if !bT: C[m,n] = sum_k A[m,k] * B[n,k]   (B row-major (Nm x K), i.e. A @ B^T)
//   if  bT: C[m,n] = sum_k A[m,k] * B[k,n]   (B row-major (K x Nm),  i.e. A @ B)
// Epilogue: + bias[n] + add0[m,n] + add1[m,n], optional tanh, stores f32 and/or bf16.
// Requires M%128==0, Nm%128==0, K%64==0.
__global__ __launch_bounds__(256)
void gemm_bf16_wmma(const unsigned short* __restrict__ A,
                    const unsigned short* __restrict__ B, int bT,
                    const float* __restrict__ bias,
                    const float* __restrict__ add0,
                    const float* __restrict__ add1,
                    float* __restrict__ outF,
                    unsigned short* __restrict__ outB,
                    int M, int Nm, int K, int doTanh)
{
  // double-buffered tiles
  __shared__ __attribute__((aligned(16))) unsigned short As[2 * BM * LDSS];
  __shared__ __attribute__((aligned(16))) unsigned short Bs[2 * BN * LDSS];

  const int tid  = threadIdx.x;
  const int lane = tid & 31;
  const int wv   = tid >> 5;       // 0..7
  const int wrow = wv & 1;         // 2 waves along M
  const int wcol = wv >> 1;        // 4 waves along N
  const int m0 = blockIdx.y * BM;
  const int n0 = blockIdx.x * BN;

  const v8f vzero = {0.f,0.f,0.f,0.f,0.f,0.f,0.f,0.f};
  v8f acc[4][2];
  #pragma unroll
  for (int mi = 0; mi < 4; ++mi)
    #pragma unroll
    for (int ni = 0; ni < 2; ++ni)
      acc[mi][ni] = vzero;

  const int rA      = lane & 15;           // fragment row within 16
  const int kb_lane = (lane >> 4) * 8;     // K sub-chunk base: 0 / 8

  stage_ab(As, Bs, A, B, bT, m0, n0, 0, K, Nm, tid);

  int cb = 0;
  for (int kt = 0; kt < K; kt += BK) {
#if USE_ASYNC_LDS
    __builtin_amdgcn_s_wait_asynccnt(0);
#endif
    __syncthreads();   // staged buffer 'cb' now visible to all waves

    unsigned short* asC = As + cb * (BM * LDSS);
    unsigned short* bsC = Bs + cb * (BN * LDSS);

    // kick off next tile into the alternate buffer
    if (kt + BK < K)
      stage_ab(As + (cb ^ 1) * (BM * LDSS), Bs + (cb ^ 1) * (BN * LDSS),
               A, B, bT, m0, n0, kt + BK, K, Nm, tid);

    // warm L2 for the tile after next (global_prefetch_b8)
    if (kt + 2 * BK < K) {
      __builtin_prefetch(&A[(size_t)(m0 + (tid & 127)) * K + kt + 2 * BK], 0, 1);
      if (!bT)
        __builtin_prefetch(&B[(size_t)(n0 + (tid & 127)) * K + kt + 2 * BK], 0, 1);
      else
        __builtin_prefetch(&B[(size_t)(kt + 2 * BK + (tid & 63)) * Nm + n0], 0, 1);
    }

    // 2 K-sub-chunks of 32: fragments per CDNA5 16-bit A/B layout
    #pragma unroll
    for (int s = 0; s < 2; ++s) {
      const int kb = s * 32 + kb_lane;
      v16bf afrag[4], bfrag[2];
      #pragma unroll
      for (int mi = 0; mi < 4; ++mi) {
        const unsigned short* p = &asC[(wrow * 64 + mi * 16 + rA) * LDSS];
        v8bf lo = *(const v8bf*)(p + kb);          // K = kb..kb+7
        v8bf hi = *(const v8bf*)(p + kb + 16);     // K = kb+16..kb+23
        afrag[mi] = __builtin_shufflevector(lo, hi, 0,1,2,3,4,5,6,7,
                                                    8,9,10,11,12,13,14,15);
      }
      #pragma unroll
      for (int ni = 0; ni < 2; ++ni) {
        const unsigned short* p = &bsC[(wcol * 32 + ni * 16 + rA) * LDSS];
        v8bf lo = *(const v8bf*)(p + kb);
        v8bf hi = *(const v8bf*)(p + kb + 16);
        bfrag[ni] = __builtin_shufflevector(lo, hi, 0,1,2,3,4,5,6,7,
                                                    8,9,10,11,12,13,14,15);
      }
      #pragma unroll
      for (int mi = 0; mi < 4; ++mi)
        #pragma unroll
        for (int ni = 0; ni < 2; ++ni)
          acc[mi][ni] = __builtin_amdgcn_wmma_f32_16x16x32_bf16(
              false, afrag[mi], false, bfrag[ni],
              (short)0, acc[mi][ni], false, false);
    }
    cb ^= 1;
  }

  // ---- epilogue ----
  #pragma unroll
  for (int mi = 0; mi < 4; ++mi) {
    #pragma unroll
    for (int ni = 0; ni < 2; ++ni) {
      int ncol = n0 + wcol * 32 + ni * 16 + (lane & 15);
      #pragma unroll
      for (int r = 0; r < 8; ++r) {
        int mrow = m0 + wrow * 64 + mi * 16 + ((lane < 16) ? r : r + 8);
        float o = acc[mi][ni][r];
        if (bias) o += bias[ncol];
        size_t off = (size_t)mrow * Nm + ncol;
        if (add0) o += add0[off];
        if (add1) o += add1[off];
        if (doTanh) o = tanhf(o);
        if (outF) outF[off] = o;
        if (outB) outB[off] = f2bf(o);
      }
    }
  }
}

// q/k projections: output K-dim is only 16 -> plain VALU kernel.
__global__ __launch_bounds__(256)
void qk_kernel(const unsigned short* __restrict__ hid,
               const float* __restrict__ qW, const float* __restrict__ qb,
               const float* __restrict__ kW, const float* __restrict__ kb,
               float* __restrict__ qo, float* __restrict__ ko)
{
  int gid = blockIdx.x * blockDim.x + threadIdx.x;   // 0 .. 2*N_AG*QKD-1
  int which = gid >> 15;                             // N_AG*QKD = 32768
  int m = (gid & 32767) >> 4;
  int j = gid & 15;
  const float* W = which ? kW : qW;
  const float* b = which ? kb : qb;
  float* o = which ? ko : qo;
  const unsigned short* h = hid + (size_t)m * HID;
  const float* w = W + (size_t)j * HID;
  float s = 0.f;
  for (int k = 0; k < HID; k += 4) {
    s += bf2f(h[k])     * w[k];
    s += bf2f(h[k + 1]) * w[k + 1];
    s += bf2f(h[k + 2]) * w[k + 2];
    s += bf2f(h[k + 3]) * w[k + 3];
  }
  o[(size_t)m * QKD + j] = s + b[j];
}

// One block per row: scores = (q @ k^T) * inv_scale, softmax, emit bf16 probs.
__global__ __launch_bounds__(256)
void attn_kernel(const float* __restrict__ q, const float* __restrict__ k,
                 unsigned short* __restrict__ attn, float inv_scale)
{
  __shared__ float sc[N_AG];
  __shared__ float red[256];
  int m = blockIdx.x, tid = threadIdx.x;
  float qv[QKD];
  #pragma unroll
  for (int j = 0; j < QKD; ++j) qv[j] = q[(size_t)m * QKD + j];

  float lmax = -1e30f;
  for (int c = tid; c < N_AG; c += 256) {
    const float* kr = k + (size_t)c * QKD;
    float s = 0.f;
    #pragma unroll
    for (int j = 0; j < QKD; ++j) s += qv[j] * kr[j];
    s *= inv_scale;
    sc[c] = s;
    lmax = fmaxf(lmax, s);
  }
  red[tid] = lmax; __syncthreads();
  for (int off = 128; off > 0; off >>= 1) {
    if (tid < off) red[tid] = fmaxf(red[tid], red[tid + off]);
    __syncthreads();
  }
  float gmax = red[0]; __syncthreads();

  float lsum = 0.f;
  for (int c = tid; c < N_AG; c += 256) {
    float e = __expf(sc[c] - gmax);
    sc[c] = e; lsum += e;
  }
  red[tid] = lsum; __syncthreads();
  for (int off = 128; off > 0; off >>= 1) {
    if (tid < off) red[tid] += red[tid + off];
    __syncthreads();
  }
  float inv = 1.f / red[0];
  for (int c = tid; c < N_AG; c += 256)
    attn[(size_t)m * N_AG + c] = f2bf(sc[c] * inv);
}

// head + log_softmax (wave 0: 32 lanes = 32 actions) and value head (wave 1).
__global__ __launch_bounds__(64)
void head_kernel(const unsigned short* __restrict__ hid,
                 const float* __restrict__ hW, const float* __restrict__ hb,
                 const float* __restrict__ vW, const float* __restrict__ vb,
                 float* __restrict__ out)
{
  __shared__ unsigned short hrow[HID];
  int m = blockIdx.x, tid = threadIdx.x;
  for (int i = tid; i < HID; i += 64) hrow[i] = hid[(size_t)m * HID + i];
  __syncthreads();
  int lane = tid & 31, wvid = tid >> 5;
  if (wvid == 0) {
    const float* w = hW + (size_t)lane * HID;
    float s = 0.f;
    for (int kk = 0; kk < HID; ++kk) s += bf2f(hrow[kk]) * w[kk];
    s += hb[lane];
    float mx = s;
    #pragma unroll
    for (int off = 16; off > 0; off >>= 1) mx = fmaxf(mx, __shfl_xor(mx, off, 32));
    float e = __expf(s - mx);
    float sum = e;
    #pragma unroll
    for (int off = 16; off > 0; off >>= 1) sum += __shfl_xor(sum, off, 32);
    out[(size_t)m * NACT + lane] = s - mx - __logf(sum);
  } else if (lane == 0) {
    float s = 0.f;
    for (int kk = 0; kk < HID; ++kk) s += bf2f(hrow[kk]) * vW[kk];
    out[(size_t)N_AG * NACT + m] = s + vb[0];
  }
}

extern "C" void kernel_launch(void* const* d_in, const int* in_sizes, int n_in,
                              void* d_out, int out_size, void* d_ws, size_t ws_size,
                              hipStream_t stream)
{
  (void)in_sizes; (void)n_in; (void)out_size; (void)ws_size;
  const float* x      = (const float*)d_in[0];
  const float* enc_W  = (const float*)d_in[1];
  const float* enc_b  = (const float*)d_in[2];
  const float* q_W    = (const float*)d_in[3];
  const float* q_b    = (const float*)d_in[4];
  const float* k_W    = (const float*)d_in[5];
  const float* k_b    = (const float*)d_in[6];
  const float* v_W    = (const float*)d_in[7];
  const float* v_b    = (const float*)d_in[8];
  const float* f_W    = (const float*)d_in[9];
  const float* f_b    = (const float*)d_in[10];
  const float* c_W    = (const float*)d_in[11];
  const float* c_b    = (const float*)d_in[12];
  const float* head_W = (const float*)d_in[13];
  const float* head_b = (const float*)d_in[14];
  const float* val_W  = (const float*)d_in[15];
  const float* val_b  = (const float*)d_in[16];
  float* out = (float*)d_out;

  char* ws = (char*)d_ws;
  size_t off = 0;
  auto alloc = [&](size_t bytes) -> char* {
    char* p = ws + off;
    off += (bytes + 255) & ~(size_t)255;
    return p;
  };
  unsigned short* x_bf    = (unsigned short*)alloc((size_t)N_AG * OBS * 2);
  unsigned short* encW_bf = (unsigned short*)alloc((size_t)HID * OBS * 2);
  unsigned short* vW_bf   = (unsigned short*)alloc((size_t)HID * HID * 2);
  unsigned short* fW_bf   = (unsigned short*)alloc((size_t)PASSES * HID * HID * 2);
  unsigned short* cW_bf   = (unsigned short*)alloc((size_t)PASSES * HID * HID * 2);
  float*          h0_f    = (float*)alloc((size_t)N_AG * HID * 4);
  unsigned short* hidA    = (unsigned short*)alloc((size_t)N_AG * HID * 2);
  unsigned short* hidB    = (unsigned short*)alloc((size_t)N_AG * HID * 2);
  unsigned short* v_bf    = (unsigned short*)alloc((size_t)N_AG * HID * 2);
  unsigned short* comm_bf = (unsigned short*)alloc((size_t)N_AG * HID * 2);
  unsigned short* attn_bf = (unsigned short*)alloc((size_t)N_AG * N_AG * 2);
  float*          c_f     = (float*)alloc((size_t)N_AG * HID * 4);
  float*          q_f     = (float*)alloc((size_t)N_AG * QKD * 4);
  float*          k_f     = (float*)alloc((size_t)N_AG * QKD * 4);

  auto cvt = [&](const float* in, unsigned short* o, size_t n) {
    int blocks = (int)((n + 2047) / 2048);
    if (blocks > 8192) blocks = 8192;
    cvt_bf16_kernel<<<dim3(blocks), dim3(256), 0, stream>>>(in, o, n);
  };
  auto gemm = [&](const unsigned short* A, const unsigned short* B, int bT,
                  const float* bias, const float* a0, const float* a1,
                  float* oF, unsigned short* oB, int M, int Nm, int K, int th) {
    dim3 g(Nm / BN, M / BM);
    gemm_bf16_wmma<<<g, dim3(256), 0, stream>>>(A, B, bT, bias, a0, a1,
                                                oF, oB, M, Nm, K, th);
  };

  // one-time (per launch) bf16 conversions of GEMM operands
  cvt(x,     x_bf,    (size_t)N_AG * OBS);
  cvt(enc_W, encW_bf, (size_t)HID * OBS);
  cvt(v_W,   vW_bf,   (size_t)HID * HID);
  cvt(f_W,   fW_bf,   (size_t)PASSES * HID * HID);
  cvt(c_W,   cW_bf,   (size_t)PASSES * HID * HID);

  // h0 = tanh(x @ enc_W^T + enc_b) -> h0 (f32 residual) and hid (bf16)
  gemm(x_bf, encW_bf, 0, enc_b, nullptr, nullptr, h0_f, hidA, N_AG, HID, OBS, 1);

  unsigned short* cur = hidA;
  unsigned short* nxt = hidB;
  const float inv_scale = 1.0f / sqrtf((float)HID);

  for (int p = 0; p < PASSES; ++p) {
    // q,k projections (K-dim 16 -> VALU kernel)
    qk_kernel<<<dim3((2 * N_AG * QKD) / 256), dim3(256), 0, stream>>>(
        cur, q_W, q_b, k_W, k_b, q_f, k_f);
    // softmax((q k^T)/sqrt(H)) -> bf16 attention
    attn_kernel<<<dim3(N_AG), dim3(256), 0, stream>>>(q_f, k_f, attn_bf, inv_scale);
    // v = hid @ v_W^T + v_b           (bf16 out, consumed as non-transposed B)
    gemm(cur, vW_bf, 0, v_b, nullptr, nullptr, nullptr, v_bf, N_AG, HID, HID, 0);
    // comm = attn @ v                 (bT: B is K x Nm row-major)
    gemm(attn_bf, v_bf, 1, nullptr, nullptr, nullptr, nullptr, comm_bf,
         N_AG, HID, N_AG, 0);
    // c = comm @ c_W[p]^T + c_b[p]    (f32 out for residual)
    gemm(comm_bf, cW_bf + (size_t)p * HID * HID, 0, c_b + (size_t)p * HID,
         nullptr, nullptr, c_f, nullptr, N_AG, HID, HID, 0);
    // hid' = tanh(h0 + hid @ f_W[p]^T + f_b[p] + c)   (double-buffered)
    gemm(cur, fW_bf + (size_t)p * HID * HID, 0, f_b + (size_t)p * HID,
         h0_f, c_f, nullptr, nxt, N_AG, HID, HID, 1);
    unsigned short* t = cur; cur = nxt; nxt = t;
  }

  // action log-softmax + value head
  head_kernel<<<dim3(N_AG), dim3(64), 0, stream>>>(
      cur, head_W, head_b, val_W, val_b, out);
}
